// GPTQLinear_38208029065299
// MI455X (gfx1250) — matrix operand order repaired and checked by reading, not compiled, not run on hardware
//
#include <hip/hip_runtime.h>
#include <hip/hip_bf16.h>

#define IN_F   4096
#define OUT_F  11008
#define BATCH  16
#define NGROUP 32
#define GSIZE  128

typedef __attribute__((ext_vector_type(16))) _Float16    v16h;
typedef __attribute__((ext_vector_type(8)))  float       v8f;
typedef __attribute__((ext_vector_type(8)))  unsigned int v8u;

// ---- workspace layout ----
#define APACK_BYTES (BATCH * IN_F * 2)          // 131072 : x in f16, A-fragment order
#define XSUM_OFF    (APACK_BYTES)               // 131072
#define XSUM_BYTES  (BATCH * NGROUP * 4)        // 2048   : per (batch,group) sums of f16(x)
#define PART_OFF    (XSUM_OFF + XSUM_BYTES)     // 133120
#define PART_ELEMS  (BATCH * OUT_F)             // 176128 f32 per K-slice

// Pack x (f32, row-major [16,4096]) into this code's 16x16x32 f16 A-fragment order (contiguous K, B uses the same):
// chunk c (K0=32c): 32 lanes x 16 halves. Lanes 0-15: M=lane, K = K0 + {0..15};
// lanes 16-31: M=lane-16, K = K0 + {16..31}.  One wave then loads its A with
// two coalesced b128 loads per chunk.
__global__ void GPTQ_pack_x(const float* __restrict__ x, _Float16* __restrict__ apack) {
    int t = blockIdx.x * 256 + threadIdx.x;
    if (t >= BATCH * IN_F) return;
    int c    = t >> 9;          // 512 halves per chunk (32 lanes * 16)
    int r    = t & 511;
    int lane = r >> 4;
    int h    = r & 15;
    int M    = lane & 15;
    int K    = c * 32 + ((lane < 16) ? 0 : 16) + h;
    apack[t] = (_Float16)x[M * IN_F + K];
}

// xsum[b,g] = sum over group g of f16(x[b,k])  (f16 to match WMMA A operand exactly)
__global__ void GPTQ_xsum(const float* __restrict__ x, float* __restrict__ xsum) {
    int t = blockIdx.x * 256 + threadIdx.x;
    if (t >= BATCH * NGROUP) return;
    int b = t >> 5, g = t & 31;
    const float* p = x + (size_t)b * IN_F + g * GSIZE;
    float s = 0.f;
    for (int k = 0; k < GSIZE; ++k) s += (float)(_Float16)p[k];
    xsum[t] = s;
}

// One wave = one 16-row (all of batch) x 16-col output tile over one K-slice.
// B operand is built from raw int4 nibbles via the 1024-bias magic number:
// f16 bits (0x6400 | n) == 1024 + n exactly for n in [0,16).
__global__ __launch_bounds__(32) void GPTQ_gemm(
    const int* __restrict__ qw,        // [OUT_F, IN_F/2] one packed byte per int32
    const float* __restrict__ scales,  // [OUT_F, NGROUP]
    const _Float16* __restrict__ apack,
    float* __restrict__ partial,       // [kSplit][BATCH][OUT_F]
    int groupsPerSlice) {
    const int lane  = threadIdx.x;       // 0..31, wave32
    const int tile  = blockIdx.x;        // 688 tiles of 16 output features
    const int slice = blockIdx.y;        // K-slice
    const int o     = tile * 16 + (lane & 15);
    const int hi    = lane >> 4;         // 0: K-low half, 1: K-high half
    const int g0    = slice * groupsPerSlice;
    const int g1    = g0 + groupsPerSlice;
    const size_t rowDw = (size_t)o * (IN_F / 2);   // dword (=byte) row pitch

    v8f acc = {0.f,0.f,0.f,0.f,0.f,0.f,0.f,0.f};

    for (int g = g0; g < g1; ++g) {
        float s = scales[o * NGROUP + g];
        if (g + 1 < g1)  // stream-prefetch next group's 64B segment of this row
            __builtin_prefetch((const char*)(qw + rowDw + (size_t)(g + 1) * 64 + hi * 8), 0, 0);

        v8f cg = {0.f,0.f,0.f,0.f,0.f,0.f,0.f,0.f};
#pragma unroll
        for (int cc = 0; cc < 4; ++cc) {          // 4 chunks of K=32 per group
            const int cG = g * 4 + cc;
            // A fragment: 32 contiguous bytes per lane from packed buffer (L2 resident)
            const uint4* pa = (const uint4*)(apack + ((size_t)cG * 32 + lane) * 16);
            uint4 a0 = pa[0], a1 = pa[1];
            v8u au = {a0.x, a0.y, a0.z, a0.w, a1.x, a1.y, a1.z, a1.w};
            v16h av = __builtin_bit_cast(v16h, au);
            // B fragment: lane holds 16 consecutive K nibbles of row o
            //  -> 8 int32 (low byte each) = two b128 loads
            const uint4* pb = (const uint4*)(qw + rowDw + (size_t)cG * 16 + hi * 8);
            uint4 w0 = pb[0], w1 = pb[1];
            unsigned wj[8] = {w0.x, w0.y, w0.z, w0.w, w1.x, w1.y, w1.z, w1.w};
            v8u bu;
#pragma unroll
            for (int j = 0; j < 8; ++j)   // (lo,hi) nibble -> half2(1024+q_even, 1024+q_odd)
                bu[j] = (((wj[j] | (wj[j] << 12)) & 0x000F000Fu) | 0x64006400u);
            v16h bv = __builtin_bit_cast(v16h, bu);

            cg = __builtin_amdgcn_wmma_f32_16x16x32_f16(
                     false, av, false, bv, (short)0, cg, false, false);
        }
#pragma unroll
        for (int r = 0; r < 8; ++r) acc[r] = fmaf(s, cg[r], acc[r]);  // per-group scale
    }

    // D layout: lanes 0-15 -> M=r, lanes 16-31 -> M=8+r; N = lane&15
    float* pp = partial + (size_t)slice * PART_ELEMS;
#pragma unroll
    for (int r = 0; r < 8; ++r) {
        int b = r + hi * 8;
        pp[(size_t)b * OUT_F + o] = acc[r];
    }
}

// out[b,o] = sum_slices partial - sum_g s[o,g]*(z[o,g]+1024)*xsum[b,g] + bias[o]
__global__ void GPTQ_reduce(const float* __restrict__ partial,
                            const float* __restrict__ scales,
                            const float* __restrict__ zeros,
                            const float* __restrict__ xsum,
                            const float* __restrict__ bias,
                            float* __restrict__ out, int kSplit) {
    int t = blockIdx.x * 256 + threadIdx.x;
    if (t >= PART_ELEMS) return;
    int b = t / OUT_F;
    int o = t - b * OUT_F;
    float r = bias[o];
    for (int s = 0; s < kSplit; ++s) r += partial[(size_t)s * PART_ELEMS + t];
    float c = 0.f;
#pragma unroll
    for (int g = 0; g < NGROUP; ++g)
        c += scales[o * NGROUP + g] * (zeros[o * NGROUP + g] + 1024.0f) * xsum[b * NGROUP + g];
    out[t] = r - c;
}

extern "C" void kernel_launch(void* const* d_in, const int* in_sizes, int n_in,
                              void* d_out, int out_size, void* d_ws, size_t ws_size,
                              hipStream_t stream) {
    const float* x      = (const float*)d_in[0];
    const int*   qw     = (const int*)  d_in[1];
    const float* scales = (const float*)d_in[2];
    const float* zeros  = (const float*)d_in[3];
    const float* bias   = (const float*)d_in[4];
    float*       out    = (float*)d_out;

    char*      ws      = (char*)d_ws;
    _Float16*  apack   = (_Float16*)ws;
    float*     xsum    = (float*)(ws + XSUM_OFF);
    float*     partial = (float*)(ws + PART_OFF);

    // K-split for parallelism (688 tiles alone under-fills the GPU); pick the
    // largest split whose f32 partials fit in the workspace. Deterministic.
    const size_t need4 = PART_OFF + 4ull * PART_ELEMS * 4;
    const size_t need2 = PART_OFF + 2ull * PART_ELEMS * 4;
    int kSplit = (ws_size >= need4) ? 4 : (ws_size >= need2) ? 2 : 1;
    int gps = NGROUP / kSplit;

    GPTQ_pack_x<<<(BATCH * IN_F + 255) / 256, 256, 0, stream>>>(x, apack);
    GPTQ_xsum  <<<(BATCH * NGROUP + 255) / 256, 256, 0, stream>>>(x, xsum);
    GPTQ_gemm  <<<dim3(OUT_F / 16, kSplit), 32, 0, stream>>>(qw, scales, apack, partial, gps);
    GPTQ_reduce<<<(PART_ELEMS + 255) / 256, 256, 0, stream>>>(partial, scales, zeros, xsum,
                                                              bias, out, kSplit);
}
